// AttnEmo_43215960932402
// MI455X (gfx1250) — matrix operand order repaired and checked by reading, not compiled, not run on hardware
//
#include <hip/hip_runtime.h>
#include <hip/hip_bf16.h>

typedef __attribute__((ext_vector_type(16))) _Float16 v16h;
typedef __attribute__((ext_vector_type(8)))  _Float16 v8h;
typedef __attribute__((ext_vector_type(8)))  float    v8f;
typedef __attribute__((ext_vector_type(4)))  float    f4;
typedef __attribute__((ext_vector_type(4)))  unsigned int u32x4;
typedef __attribute__((ext_vector_type(4)))  int      i32x4;
typedef __attribute__((ext_vector_type(8)))  int      i32x8;

#define E_DIM  512
#define SEQ    2048
#define NBATCH 8

// ---------------------------------------------------------------------------
// WMMA helper (CDNA5 gfx1250, wave32): D = A(16x32 f16) * B(32x16 f16) + C(f32)
// ---------------------------------------------------------------------------
__device__ __forceinline__ v8f wmma16(v16h a, v16h b, v8f c) {
  return __builtin_amdgcn_wmma_f32_16x16x32_f16(
      /*neg_a=*/false, a, /*neg_b=*/false, b,
      /*c_mod=*/(short)0, c, /*reuse_a=*/false, /*reuse_b=*/false);
}

__device__ __forceinline__ v16h pack16(v8h lo, v8h hi) {
  v16h a;
#pragma unroll
  for (int i = 0; i < 8; i++) { a[i] = lo[i]; a[i + 8] = hi[i]; }
  return a;
}

// A fragment from f32 row-major memory. p = row + k0 + lt*16.
// halves 0..15 <- K[kbase..kbase+15]
__device__ __forceinline__ v16h load_a_f32(const float* p) {
  f4 x0 = *(const f4*)(p);
  f4 x1 = *(const f4*)(p + 4);
  f4 x2 = *(const f4*)(p + 8);
  f4 x3 = *(const f4*)(p + 12);
  v16h a;
#pragma unroll
  for (int i = 0; i < 4; i++) {
    a[i]      = (_Float16)x0[i];
    a[4 + i]  = (_Float16)x1[i];
    a[8 + i]  = (_Float16)x2[i];
    a[12 + i] = (_Float16)x3[i];
  }
  return a;
}

// ---------------------------------------------------------------------------
// Tensor Data Mover: 1-D contiguous f16 tile, global -> LDS.
// D# layout per CDNA5 ISA.
// ---------------------------------------------------------------------------
__device__ __forceinline__ void tdm_load_contig_f16(unsigned lds_byte_off,
                                                    const _Float16* src,
                                                    unsigned nelem /* <= 65535 */)
{
  const unsigned long long ga = (unsigned long long)(uintptr_t)src;
  u32x4 g0;
  g0.x = 1u;                                           // count=1, user mode
  g0.y = lds_byte_off;                                 // lds_addr
  g0.z = (unsigned)(ga & 0xFFFFFFFFull);               // global_addr[31:0]
  g0.w = (unsigned)((ga >> 32) & 0x01FFFFFFull)        // global_addr[56:32]
       | 0x80000000u;                                  // type=2 ("image")

  i32x8 g1;
  g1[0] = 0x00010000;                                  // data_size=1 (2 bytes)
  g1[1] = (int)((nelem & 0xFFFFu) << 16);              // tensor_dim0[15:0]
  g1[2] = (int)(((nelem >> 16) & 0xFFFFu) | (1u << 16)); // dim0 hi | tensor_dim1=1
  g1[3] = (int)((nelem & 0xFFFFu) << 16);              // tile_dim0
  g1[4] = 1;                                           // tile_dim1=1, tile_dim2=0
  g1[5] = (int)nelem;                                  // tensor_dim0_stride lo
  g1[6] = 0;
  g1[7] = 0;

  i32x4 gz;
  gz[0] = 0; gz[1] = 0; gz[2] = 0; gz[3] = 0;

#if defined(__clang_major__) && (__clang_major__ >= 23)
  i32x8 g4;
#pragma unroll
  for (int i = 0; i < 8; i++) g4[i] = 0;
  __builtin_amdgcn_tensor_load_to_lds(g0, g1, gz, gz, g4, 0);
#else
  __builtin_amdgcn_tensor_load_to_lds(g0, g1, gz, gz, 0);
#endif
}

// ---------------------------------------------------------------------------
// f32 -> f16 conversion (weights; one-time, trivial traffic)
// ---------------------------------------------------------------------------
__global__ void __launch_bounds__(256)
cvt_kernel(const float* __restrict__ in, _Float16* __restrict__ out, int n)
{
  for (int i = blockIdx.x * blockDim.x + threadIdx.x; i < n;
       i += gridDim.x * blockDim.x)
    out[i] = (_Float16)in[i];
}

// ---------------------------------------------------------------------------
// Projection: Y = X @ W^T + bias. X is f32 (converted in-register); W is
// pre-converted f16 [out,in] row-major. Block = 64(M) x 64(N), 4 waves.
// transpose==0: Y row-major [rows, E]; transpose==1: Y^T as [B, E, T].
// ---------------------------------------------------------------------------
__global__ void __launch_bounds__(128)
proj_kernel(const float* __restrict__ X, const _Float16* __restrict__ Wh,
            const float* __restrict__ bias, _Float16* __restrict__ Y,
            const int transpose)
{
  const int lane  = threadIdx.x & 31;
  const int w     = threadIdx.x >> 5;
  const int lh    = lane & 15;
  const int lt    = lane >> 4;
  const int mtile = blockIdx.x >> 3;
  const int ntile = blockIdx.x & 7;
  const int mbase = mtile * 64 + w * 16;
  const int nbase = ntile * 64;

  v8f c[4];
#pragma unroll
  for (int n = 0; n < 4; n++)
#pragma unroll
    for (int i = 0; i < 8; i++) c[n][i] = 0.f;

  const float* arow = X + (size_t)(mbase + lh) * E_DIM;
#pragma unroll 4
  for (int k0 = 0; k0 < E_DIM; k0 += 32) {
    v16h a = load_a_f32(arow + k0 + lt * 16);
#pragma unroll
    for (int n = 0; n < 4; n++) {
      const int nn = nbase + n * 16 + lh;
      v16h b = *(const v16h*)(Wh + (size_t)nn * E_DIM + k0 + lt * 16);
      c[n] = wmma16(a, b, c[n]);
    }
  }
#pragma unroll
  for (int n = 0; n < 4; n++) {
    const float bv = bias[nbase + n * 16 + lh];
#pragma unroll
    for (int r = 0; r < 8; r++) c[n][r] += bv;
  }

  if (!transpose) {
#pragma unroll
    for (int n = 0; n < 4; n++) {
      const int col = nbase + n * 16 + lh;
#pragma unroll
      for (int r = 0; r < 8; r++) {
        const int row = mbase + lt * 8 + r;
        Y[(size_t)row * E_DIM + col] = (_Float16)c[n][r];
      }
    }
  } else {
    const int bb = mbase / SEQ;
    const int tl = mbase % SEQ;
#pragma unroll
    for (int n = 0; n < 4; n++) {
      const int e = nbase + n * 16 + lh;
      v8h pk;
#pragma unroll
      for (int r = 0; r < 8; r++) pk[r] = (_Float16)c[n][r];
      *(v8h*)(Y + ((size_t)bb * E_DIM + e) * SEQ + tl + lt * 8) = pk;
    }
  }
}

// ---------------------------------------------------------------------------
// Flash-style attention: per block 16 S-rows; stream T in chunks of 64 with
// online softmax; 4 waves split the 64 T-cols for QK^T and the 512 E-cols
// (128 each) for P@V. Q tile staged into LDS via the Tensor Data Mover.
// ---------------------------------------------------------------------------
__global__ void __launch_bounds__(128)
attn_kernel(const _Float16* __restrict__ Qh, const _Float16* __restrict__ Kh,
            const _Float16* __restrict__ VhT, const unsigned char* __restrict__ mask,
            _Float16* __restrict__ Ch)
{
  __shared__ _Float16 ldsQ[16 * E_DIM];   // Q tile, row-major [16][512]
  __shared__ _Float16 ldsP[16 * 64];      // P tile, row-major [16][64]
  __shared__ float ldsMax[4][16];
  __shared__ float ldsSum[4][16];

  const int b    = blockIdx.x >> 7;
  const int s0   = (blockIdx.x & 127) << 4;
  const int lane = threadIdx.x & 31;
  const int w    = threadIdx.x >> 5;
  const int lh   = lane & 15;
  const int lt   = lane >> 4;

  // Stage Q tile (16 rows contiguous = 16 KB) with one TDM DMA from wave 0.
  if (w == 0) {
    tdm_load_contig_f16((unsigned)(uintptr_t)(void*)ldsQ,
                        Qh + ((size_t)b * SEQ + s0) * E_DIM, 16 * E_DIM);
    __builtin_amdgcn_s_wait_tensorcnt(0);
  }
  __syncthreads();

  float m_run[8], l_run[8];
#pragma unroll
  for (int r = 0; r < 8; r++) { m_run[r] = -3.0e38f; l_run[r] = 0.f; }
  v8f O[8];
#pragma unroll
  for (int n = 0; n < 8; n++)
#pragma unroll
    for (int i = 0; i < 8; i++) O[n][i] = 0.f;

  const int tcol_local = w * 16 + lh;

  for (int t0 = 0; t0 < SEQ; t0 += 64) {
    const int tcol = t0 + tcol_local;
    const _Float16* krow = Kh + ((size_t)b * SEQ + tcol) * E_DIM;

    // Prefetch next T-block's K row and V^T row toward the WGP.
    if (t0 + 64 < SEQ) {
      __builtin_prefetch(krow + (size_t)64 * E_DIM, 0, 1);
      __builtin_prefetch(VhT + ((size_t)b * E_DIM + w * 128 + lh) * SEQ + t0 + 64, 0, 1);
    }

    // ---- QK^T: this wave computes logits[16][16] at cols [t0+w*16, +16)
    v8f cqk;
#pragma unroll
    for (int i = 0; i < 8; i++) cqk[i] = 0.f;
#pragma unroll 4
    for (int k0 = 0; k0 < E_DIM; k0 += 32) {
      v8h alo = *(const v8h*)(ldsQ + lh * E_DIM + k0 + lt * 16);
      v8h ahi = *(const v8h*)(ldsQ + lh * E_DIM + k0 + lt * 16 + 8);
      v16h a  = pack16(alo, ahi);
      v16h bf = *(const v16h*)(krow + k0 + lt * 16);
      cqk = wmma16(a, bf, cqk);
    }

    // ---- mask + per-row max (rows m = lt*8 + r, col n = lh)
    const unsigned char* mrow =
        mask + ((size_t)b * SEQ + (s0 + lt * 8)) * SEQ + tcol;
    float lg[8];
#pragma unroll
    for (int r = 0; r < 8; r++)
      lg[r] = mrow[(size_t)r * SEQ] ? -1.0e18f : cqk[r];

    float red[8];
#pragma unroll
    for (int r = 0; r < 8; r++) red[r] = lg[r];
#pragma unroll
    for (int off = 1; off < 16; off <<= 1)
#pragma unroll
      for (int r = 0; r < 8; r++)
        red[r] = fmaxf(red[r], __shfl_xor(red[r], off, 32));
    if (lh == 0) {
#pragma unroll
      for (int r = 0; r < 8; r++) ldsMax[w][lt * 8 + r] = red[r];
    }
    __syncthreads();

    float fac[8], p[8];
#pragma unroll
    for (int r = 0; r < 8; r++) {
      const int row = lt * 8 + r;
      float tm = fmaxf(fmaxf(ldsMax[0][row], ldsMax[1][row]),
                       fmaxf(ldsMax[2][row], ldsMax[3][row]));
      float mnew = fmaxf(m_run[r], tm);
      fac[r] = __expf(m_run[r] - mnew);
      m_run[r] = mnew;
      p[r] = __expf(lg[r] - mnew);
    }

    // ---- row sums of p
#pragma unroll
    for (int r = 0; r < 8; r++) red[r] = p[r];
#pragma unroll
    for (int off = 1; off < 16; off <<= 1)
#pragma unroll
      for (int r = 0; r < 8; r++) red[r] += __shfl_xor(red[r], off, 32);
    if (lh == 0) {
#pragma unroll
      for (int r = 0; r < 8; r++) ldsSum[w][lt * 8 + r] = red[r];
    }
    // write P tile (f16) into LDS for A-fragment reload
#pragma unroll
    for (int r = 0; r < 8; r++)
      ldsP[(lt * 8 + r) * 64 + tcol_local] = (_Float16)p[r];
    __syncthreads();

#pragma unroll
    for (int r = 0; r < 8; r++) {
      const int row = lt * 8 + r;
      const float ts = ldsSum[0][row] + ldsSum[1][row] +
                       ldsSum[2][row] + ldsSum[3][row];
      l_run[r] = l_run[r] * fac[r] + ts;
    }
    // rescale accumulators (per-row factor matches C-fragment row layout)
#pragma unroll
    for (int n = 0; n < 8; n++)
#pragma unroll
      for (int r = 0; r < 8; r++) O[n][r] *= fac[r];

    // ---- P @ V : wave's E slice [w*128, w*128+128), K=64 in two chunks
#pragma unroll
    for (int kc = 0; kc < 2; kc++) {
      v8h plo = *(const v8h*)(ldsP + lh * 64 + kc * 32 + lt * 16);
      v8h phi = *(const v8h*)(ldsP + lh * 64 + kc * 32 + lt * 16 + 8);
      v16h pa = pack16(plo, phi);
#pragma unroll
      for (int n = 0; n < 8; n++) {
        const int e = w * 128 + n * 16 + lh;
        v16h bf = *(const v16h*)(VhT + ((size_t)b * E_DIM + e) * SEQ +
                                 t0 + kc * 32 + lt * 16);
        O[n] = wmma16(pa, bf, O[n]);
      }
    }
    __syncthreads();
  }

  // ---- epilogue: divide by softmax denom, store contexts (f16)
  float invl[8];
#pragma unroll
  for (int r = 0; r < 8; r++) invl[r] = 1.f / l_run[r];
#pragma unroll
  for (int n = 0; n < 8; n++) {
    const int e = w * 128 + n * 16 + lh;
#pragma unroll
    for (int r = 0; r < 8; r++) {
      const int srow = s0 + lt * 8 + r;
      Ch[((size_t)b * SEQ + srow) * E_DIM + e] = (_Float16)(O[n][r] * invl[r]);
    }
  }
}

// ---------------------------------------------------------------------------
// out = enc + LayerNorm(enc + contexts @ Wo^T)
// Block = 16 S-rows; 4 waves each own a 128-wide E slice. Wo pre-converted f16.
// ---------------------------------------------------------------------------
__global__ void __launch_bounds__(128)
out_kernel(const _Float16* __restrict__ Ch, const _Float16* __restrict__ Woh,
           const float* __restrict__ enc, const float* __restrict__ gamma,
           const float* __restrict__ beta, float* __restrict__ out)
{
  __shared__ float ldsSum[4][16];
  __shared__ float ldsSq[4][16];
  const int b    = blockIdx.x >> 7;
  const int s0   = (blockIdx.x & 127) << 4;
  const int lane = threadIdx.x & 31;
  const int w    = threadIdx.x >> 5;
  const int lh   = lane & 15;
  const int lt   = lane >> 4;

  v8f c[8];
#pragma unroll
  for (int n = 0; n < 8; n++)
#pragma unroll
    for (int i = 0; i < 8; i++) c[n][i] = 0.f;

  const _Float16* arow = Ch + ((size_t)b * SEQ + s0 + lh) * E_DIM;
#pragma unroll 2
  for (int k0 = 0; k0 < E_DIM; k0 += 32) {
    v8h alo = *(const v8h*)(arow + k0 + lt * 16);
    v8h ahi = *(const v8h*)(arow + k0 + lt * 16 + 8);
    v16h a = pack16(alo, ahi);
#pragma unroll
    for (int n = 0; n < 8; n++) {
      const int nn = w * 128 + n * 16 + lh;
      v16h bf = *(const v16h*)(Woh + (size_t)nn * E_DIM + k0 + lt * 16);
      c[n] = wmma16(a, bf, c[n]);
    }
  }

  // residual add + row statistics
  float psum[8], psq[8];
#pragma unroll
  for (int r = 0; r < 8; r++) { psum[r] = 0.f; psq[r] = 0.f; }
#pragma unroll
  for (int n = 0; n < 8; n++) {
    const int e = w * 128 + n * 16 + lh;
#pragma unroll
    for (int r = 0; r < 8; r++) {
      const int srow = s0 + lt * 8 + r;
      const float ev = enc[((size_t)b * SEQ + srow) * E_DIM + e];
      const float xv = c[n][r] + ev;
      c[n][r] = xv;
      psum[r] += xv;
      psq[r]  += xv * xv;
    }
  }
#pragma unroll
  for (int off = 1; off < 16; off <<= 1)
#pragma unroll
    for (int r = 0; r < 8; r++) {
      psum[r] += __shfl_xor(psum[r], off, 32);
      psq[r]  += __shfl_xor(psq[r],  off, 32);
    }
  if (lh == 0) {
#pragma unroll
    for (int r = 0; r < 8; r++) {
      ldsSum[w][lt * 8 + r] = psum[r];
      ldsSq[w][lt * 8 + r]  = psq[r];
    }
  }
  __syncthreads();

  float mean[8], inv[8];
#pragma unroll
  for (int r = 0; r < 8; r++) {
    const int row = lt * 8 + r;
    const float s = ldsSum[0][row] + ldsSum[1][row] + ldsSum[2][row] + ldsSum[3][row];
    const float q = ldsSq[0][row]  + ldsSq[1][row]  + ldsSq[2][row]  + ldsSq[3][row];
    const float mu = s * (1.0f / 512.0f);
    float var = q * (1.0f / 512.0f) - mu * mu;
    var = fmaxf(var, 0.f);
    mean[r] = mu;
    inv[r] = 1.0f / (sqrtf(var) + 1e-6f);  // std-based LN, eps added to std
  }

#pragma unroll
  for (int n = 0; n < 8; n++) {
    const int e = w * 128 + n * 16 + lh;
    const float g = gamma[e], be = beta[e];
#pragma unroll
    for (int r = 0; r < 8; r++) {
      const int srow = s0 + lt * 8 + r;
      const size_t idx = ((size_t)b * SEQ + srow) * E_DIM + e;
      out[idx] = enc[idx] + g * (c[n][r] - mean[r]) * inv[r] + be;
    }
  }
}

// ---------------------------------------------------------------------------
extern "C" void kernel_launch(void* const* d_in, const int* in_sizes, int n_in,
                              void* d_out, int out_size, void* d_ws, size_t ws_size,
                              hipStream_t stream)
{
  const float* enc  = (const float*)d_in[0];
  const float* emo  = (const float*)d_in[1];
  const unsigned char* mask = (const unsigned char*)d_in[2];
  const float* Wq   = (const float*)d_in[3];
  const float* bq   = (const float*)d_in[4];
  const float* Wk   = (const float*)d_in[5];
  const float* bk   = (const float*)d_in[6];
  const float* Wv   = (const float*)d_in[7];
  const float* bv   = (const float*)d_in[8];
  const float* Wo   = (const float*)d_in[9];
  const float* gamma = (const float*)d_in[10];
  const float* beta  = (const float*)d_in[11];
  float* out = (float*)d_out;

  const size_t NE = (size_t)NBATCH * SEQ * E_DIM;   // 8.39M elems
  const int    NW = E_DIM * E_DIM;                  // 262144 elems
  _Float16* Qh  = (_Float16*)d_ws;                  // [B,S,E]  f16
  _Float16* Kh  = Qh + NE;                          // [B,T,E]  f16
  _Float16* VhT = Kh + NE;                          // [B,E,T]  f16 (V transposed)
  _Float16* Ch  = VhT + NE;                         // [B,S,E]  f16 contexts
  _Float16* Wqh = Ch + NE;                          // f16 weights
  _Float16* Wkh = Wqh + NW;
  _Float16* Wvh = Wkh + NW;
  _Float16* Woh = Wvh + NW;

  dim3 blk(128);
  cvt_kernel<<<dim3(512), dim3(256), 0, stream>>>(Wq, Wqh, NW);
  cvt_kernel<<<dim3(512), dim3(256), 0, stream>>>(Wk, Wkh, NW);
  cvt_kernel<<<dim3(512), dim3(256), 0, stream>>>(Wv, Wvh, NW);
  cvt_kernel<<<dim3(512), dim3(256), 0, stream>>>(Wo, Woh, NW);
  proj_kernel<<<dim3(2048), blk, 0, stream>>>(enc, Wqh, bq, Qh, 0);
  proj_kernel<<<dim3(2048), blk, 0, stream>>>(emo, Wkh, bk, Kh, 0);
  proj_kernel<<<dim3(2048), blk, 0, stream>>>(emo, Wvh, bv, VhT, 1);
  attn_kernel<<<dim3(1024), blk, 0, stream>>>(Qh, Kh, VhT, mask, Ch);
  out_kernel<<<dim3(1024), blk, 0, stream>>>(Ch, Woh, enc, gamma, beta, out);

  (void)in_sizes; (void)n_in; (void)out_size; (void)ws_size;
}
